// GNN_graph_clasif_model_28793460752816
// MI455X (gfx1250) — matrix operand order repaired and checked by reading, not compiled, not run on hardware
//
#include <hip/hip_runtime.h>

// CDNA5 / gfx1250 GCN (2-layer) + pool + head.
// Dominant cost: edge scatter (L2-resident, atomic-bound). Dense GEMMs run on
// the WMMA pipe: v_wmma_f32_16x16x32_f16 with f32 accumulators.

typedef __attribute__((ext_vector_type(16))) _Float16 v16h;
typedef __attribute__((ext_vector_type(8)))  float    v8f;

#define N_NODES   50000
#define N_EDGES   800000
#define N_FEAT    128
#define HIDDEN    64
#define N_CLASSES 10
#define N_GRAPHS  512

// ---------------- utility kernels ----------------

__global__ void k_zero(float* __restrict__ p, int n) {
  int i = blockIdx.x * blockDim.x + threadIdx.x;
  if (i < n) p[i] = 0.0f;
}

// Convert W1 [128x64] and W2 [64x64] (row-major, K x N) into transposed f16
// [N][K] so a WMMA B-fragment per lane is one contiguous 32B run.
__global__ void k_convert_w(const float* __restrict__ W1, const float* __restrict__ W2,
                            _Float16* __restrict__ w1t, _Float16* __restrict__ w2t) {
  int i = blockIdx.x * blockDim.x + threadIdx.x;
  if (i < N_FEAT * HIDDEN) {
    int k = i / HIDDEN, c = i % HIDDEN;
    w1t[c * N_FEAT + k] = (_Float16)W1[i];
  }
  int j = i - N_FEAT * HIDDEN;
  if (j >= 0 && j < HIDDEN * HIDDEN) {
    int k = j / HIDDEN, c = j % HIDDEN;
    w2t[c * HIDDEN + k] = (_Float16)W2[j];
  }
}

__global__ void k_degree(const int* __restrict__ ei, float* __restrict__ deg) {
  int e = blockIdx.x * blockDim.x + threadIdx.x;
  if (e < N_EDGES) atomicAdd(&deg[ei[N_EDGES + e]], 1.0f);
}

__global__ void k_dinv(float* __restrict__ deg) {
  int i = blockIdx.x * blockDim.x + threadIdx.x;
  if (i < N_NODES) {
    float d = deg[i] + 1.0f;   // +1 self-loop
    deg[i] = (d > 0.0f) ? rsqrtf(d) : 0.0f;
  }
}

// ---------------- WMMA GEMM: out[M x 64] = A[M x K] @ W[K x 64] ----------------
// One wave (32 threads) per 16-row tile; 4 accumulators cover all 64 output cols.
// A fragment (16-bit A 16x32): lane L -> row L%16; per-lane 16 halves
// are K = kb + 16*(L/16) + {0..15} (same K order as B).
// B fragment: col N = L%16, K = kb + 16*(L/16) + {0..15}; WT is [N][K] f16 so that
// run is one contiguous 32B load.
template <int K>
__global__ void k_gemm_wmma(const float* __restrict__ A,
                            const _Float16* __restrict__ WT,
                            float* __restrict__ out) {
  const int mt   = blockIdx.x;
  const int lane = threadIdx.x;
  const int hs   = lane >> 4;      // 0 or 1
  const int l16  = lane & 15;
  const float* arow = A + (size_t)(mt * 16 + l16) * K;

  v8f acc0 = {}, acc1 = {}, acc2 = {}, acc3 = {};

  for (int kb = 0; kb < K; kb += 32) {
    const float* ap0 = arow + kb + 16 * hs;
    const float* ap1 = arow + kb + 16 * hs + 8;
    float4 u0 = *(const float4*)(ap0);
    float4 u1 = *(const float4*)(ap0 + 4);
    float4 u2 = *(const float4*)(ap1);
    float4 u3 = *(const float4*)(ap1 + 4);
    v16h a;
    a[0]  = (_Float16)u0.x;  a[1]  = (_Float16)u0.y;
    a[2]  = (_Float16)u0.z;  a[3]  = (_Float16)u0.w;
    a[4]  = (_Float16)u1.x;  a[5]  = (_Float16)u1.y;
    a[6]  = (_Float16)u1.z;  a[7]  = (_Float16)u1.w;
    a[8]  = (_Float16)u2.x;  a[9]  = (_Float16)u2.y;
    a[10] = (_Float16)u2.z;  a[11] = (_Float16)u2.w;
    a[12] = (_Float16)u3.x;  a[13] = (_Float16)u3.y;
    a[14] = (_Float16)u3.z;  a[15] = (_Float16)u3.w;

    const _Float16* wb = WT + (size_t)l16 * K + kb + 16 * hs;  // col l16, ktile base
    v16h b0 = *(const v16h*)(wb + (size_t)0  * 16 * K);
    v16h b1 = *(const v16h*)(wb + (size_t)1  * 16 * K);
    v16h b2 = *(const v16h*)(wb + (size_t)2  * 16 * K);
    v16h b3 = *(const v16h*)(wb + (size_t)3  * 16 * K);

    acc0 = __builtin_amdgcn_wmma_f32_16x16x32_f16(false, a, false, b0, (short)0, acc0, false, false);
    acc1 = __builtin_amdgcn_wmma_f32_16x16x32_f16(false, a, false, b1, (short)0, acc1, false, false);
    acc2 = __builtin_amdgcn_wmma_f32_16x16x32_f16(false, a, false, b2, (short)0, acc2, false, false);
    acc3 = __builtin_amdgcn_wmma_f32_16x16x32_f16(false, a, false, b3, (short)0, acc3, false, false);
  }

  // C/D layout: VGPR j -> row M = j + 8*(lane/16), col N = lane%16 (+16*nt)
  float* obase = out + (size_t)(mt * 16 + 8 * hs) * HIDDEN + l16;
#pragma unroll
  for (int j = 0; j < 8; ++j) {
    obase[(size_t)j * HIDDEN + 0]  = acc0[j];
    obase[(size_t)j * HIDDEN + 16] = acc1[j];
    obase[(size_t)j * HIDDEN + 32] = acc2[j];
    obase[(size_t)j * HIDDEN + 48] = acc3[j];
  }
}

// ---------------- edge scatter: acc[col] += t[row] * dinv[row]*dinv[col] ----------------
// 16 threads per edge, float4 per thread (contiguous 256B gather per edge).
__global__ void k_scatter(const int* __restrict__ ei, const float* __restrict__ dinv,
                          const float* __restrict__ t, float* __restrict__ acc) {
  int idx = blockIdx.x * blockDim.x + threadIdx.x;
  if (idx >= N_EDGES * 16) return;
  int e  = idx >> 4;
  int c4 = (idx & 15) * 4;
  int r  = ei[e];
  int cl = ei[N_EDGES + e];
  float nrm = dinv[r] * dinv[cl];
  float4 v = *(const float4*)(t + (size_t)r * HIDDEN + c4);
  float* dst = acc + (size_t)cl * HIDDEN + c4;
  atomicAdd(dst + 0, v.x * nrm);
  atomicAdd(dst + 1, v.y * nrm);
  atomicAdd(dst + 2, v.z * nrm);
  atomicAdd(dst + 3, v.w * nrm);
}

// epilogue: acc = relu(acc + t*dinv^2 + b)   (self-loop term fused in)
__global__ void k_epilogue(const float* __restrict__ t, const float* __restrict__ dinv,
                           const float* __restrict__ b, float* __restrict__ acc) {
  int idx = blockIdx.x * blockDim.x + threadIdx.x;
  if (idx >= N_NODES * HIDDEN) return;
  int i = idx >> 6, c = idx & 63;
  float di = dinv[i];
  float v = acc[idx] + t[idx] * di * di + b[c];
  acc[idx] = v > 0.0f ? v : 0.0f;
}

__global__ void k_pool(const float* __restrict__ h, const int* __restrict__ batch,
                       float* __restrict__ g) {
  int idx = blockIdx.x * blockDim.x + threadIdx.x;
  if (idx >= N_NODES * HIDDEN) return;
  int i = idx >> 6, c = idx & 63;
  atomicAdd(&g[(size_t)batch[i] * HIDDEN + c], h[idx]);
}

__global__ void k_head(const float* __restrict__ g, const float* __restrict__ Wf,
                       const float* __restrict__ bf, float* __restrict__ out) {
  int gi = blockIdx.x * blockDim.x + threadIdx.x;
  if (gi >= N_GRAPHS) return;
  float lg[N_CLASSES];
#pragma unroll
  for (int c = 0; c < N_CLASSES; ++c) lg[c] = bf[c];
  for (int k = 0; k < HIDDEN; ++k) {
    float gv = g[(size_t)gi * HIDDEN + k];
#pragma unroll
    for (int c = 0; c < N_CLASSES; ++c) lg[c] += gv * Wf[k * N_CLASSES + c];
  }
  float m = lg[0];
#pragma unroll
  for (int c = 1; c < N_CLASSES; ++c) m = fmaxf(m, lg[c]);
  float s = 0.0f;
#pragma unroll
  for (int c = 0; c < N_CLASSES; ++c) s += expf(lg[c] - m);
  float lse = m + logf(s);
#pragma unroll
  for (int c = 0; c < N_CLASSES; ++c) out[(size_t)gi * N_CLASSES + c] = lg[c] - lse;
}

// ---------------- launcher ----------------

extern "C" void kernel_launch(void* const* d_in, const int* in_sizes, int n_in,
                              void* d_out, int out_size, void* d_ws, size_t ws_size,
                              hipStream_t stream) {
  (void)in_sizes; (void)n_in; (void)out_size; (void)ws_size;

  const float* x     = (const float*)d_in[0];
  const int*   ei    = (const int*)d_in[1];   // [2, E] flat: row then col
  const int*   batch = (const int*)d_in[2];
  const float* W1    = (const float*)d_in[3];
  const float* b1    = (const float*)d_in[4];
  const float* W2    = (const float*)d_in[5];
  const float* b2    = (const float*)d_in[6];
  const float* Wf    = (const float*)d_in[7];
  const float* bf    = (const float*)d_in[8];
  float*       out   = (float*)d_out;

  // workspace layout (all 256B aligned)
  char* ws = (char*)d_ws;
  float*    B0   = (float*)(ws + 0);                 // 50000*64*4 = 12,800,000
  float*    B1   = (float*)(ws + 12800000);          // 12,800,000
  float*    dinv = (float*)(ws + 25600000);          // 200,000  (deg -> dinv in place)
  _Float16* w1t  = (_Float16*)(ws + 25800192);       // 64*128*2 = 16,384
  _Float16* w2t  = (_Float16*)(ws + 25816576);       // 64*64*2  = 8,192
  float*    gp   = (float*)(ws + 25824768);          // 512*64*4 = 131,072

  const int NH = N_NODES * HIDDEN;       // 3,200,000
  const int ESL = N_EDGES * 16;          // 12,800,000 scatter lanes

  // init
  k_zero<<<(N_NODES + 255) / 256, 256, 0, stream>>>(dinv, N_NODES);
  k_zero<<<(NH + 255) / 256, 256, 0, stream>>>(B1, NH);
  k_zero<<<(N_GRAPHS * HIDDEN + 255) / 256, 256, 0, stream>>>(gp, N_GRAPHS * HIDDEN);
  k_convert_w<<<(N_FEAT * HIDDEN + HIDDEN * HIDDEN + 255) / 256, 256, 0, stream>>>(W1, W2, w1t, w2t);

  // gcn normalization
  k_degree<<<(N_EDGES + 255) / 256, 256, 0, stream>>>(ei, dinv);
  k_dinv<<<(N_NODES + 255) / 256, 256, 0, stream>>>(dinv);

  // layer 1: B0 = x@W1 ; B1 = scatter(B0) ; B1 = relu(B1 + B0*dinv^2 + b1)
  k_gemm_wmma<N_FEAT><<<N_NODES / 16, 32, 0, stream>>>(x, w1t, B0);
  k_scatter<<<(ESL + 255) / 256, 256, 0, stream>>>(ei, dinv, B0, B1);
  k_epilogue<<<(NH + 255) / 256, 256, 0, stream>>>(B0, dinv, b1, B1);

  // layer 2: B0 = h1@W2 ; B1 (re-zeroed) accumulates ; h2 -> B1
  k_gemm_wmma<HIDDEN><<<N_NODES / 16, 32, 0, stream>>>(B1, w2t, B0);
  k_zero<<<(NH + 255) / 256, 256, 0, stream>>>(B1, NH);
  k_scatter<<<(ESL + 255) / 256, 256, 0, stream>>>(ei, dinv, B0, B1);
  k_epilogue<<<(NH + 255) / 256, 256, 0, stream>>>(B0, dinv, b2, B1);

  // pool + head
  k_pool<<<(NH + 255) / 256, 256, 0, stream>>>(B1, batch, gp);
  k_head<<<(N_GRAPHS + 63) / 64, 64, 0, stream>>>(gp, Wf, bf, out);
}
